// HAN_28037546508936
// MI455X (gfx1250) — matrix operand order repaired and checked by reading, not compiled, not run on hardware
//
#include <hip/hip_runtime.h>
#include <hip/hip_bf16.h>
#include <math.h>

// ---------------------------------------------------------------- constants
#define NM 100000
#define ND 50000
#define NA 200000
#define FIN 256
#define HID 128
#define NCLS 16
#define E_MD 200000
#define E_MA 600000
#define NEG_SLOPE 0.2f

typedef unsigned short ushortT;
typedef __attribute__((ext_vector_type(16))) __bf16 v16bf;
typedef __attribute__((ext_vector_type(2)))  __bf16 v2bf;
typedef __attribute__((ext_vector_type(8)))  float  v8f;

union FragB16 { v16bf v; uint4 q[2]; };

__device__ __forceinline__ ushortT f2bf(float f) {
    unsigned u = __float_as_uint(f);
    unsigned r = ((u >> 16) & 1u) + 0x7FFFu;          // round-to-nearest-even
    return (ushortT)((u + r) >> 16);
}
__device__ __forceinline__ unsigned pk2bf(float lo, float hi) {
#if __has_builtin(__builtin_amdgcn_cvt_pk_bf16_f32)
    union { v2bf v; unsigned u; } c;
    c.v = __builtin_amdgcn_cvt_pk_bf16_f32(lo, hi);   // single v_cvt_pk_bf16_f32
    return c.u;
#else
    return (unsigned)f2bf(lo) | ((unsigned)f2bf(hi) << 16);
#endif
}
__device__ __forceinline__ unsigned encOrd(float f) { // order-preserving float->uint
    unsigned u = __float_as_uint(f);
    return (u & 0x80000000u) ? ~u : (u | 0x80000000u);
}
__device__ __forceinline__ float decOrd(unsigned u) {
    return __uint_as_float((u & 0x80000000u) ? (u & 0x7FFFFFFFu) : ~u);
}

// ---------------------------------------------------------------- utilities
__global__ void han_fill_f32(float* p, float v, long n) {
    long i = (long)blockIdx.x * blockDim.x + threadIdx.x;
    if (i < n) p[i] = v;
}

// W [K][Nn] f32  ->  Wt [Nn][K] bf16 (column-major for B fragments)
__global__ void han_transcvt(const float* W, ushortT* Wt, int K, int Nn) {
    int i = blockIdx.x * blockDim.x + threadIdx.x;
    if (i >= K * Nn) return;
    int k = i / Nn, n = i % Nn;
    Wt[n * K + k] = f2bf(W[(long)k * Nn + n]);
}

// ---------------------------------------------------------------- WMMA GEMM (shared core)
// X f32 [M][K], Wt bf16 [128][K] (col-major of W).
// TANHSUM=false: Out[m][c] = (X@W)[m][c] + bias[c]
// TANHSUM=true : colsum[c] += sum_m tanh((X@W)[m][c] + bias[c])
#define A_LD 40                      // A LDS row stride (elems): 80B rows, skewed banks
template <int K, bool TANHSUM>
__global__ __launch_bounds__(256) void han_gemm(
    const float* __restrict__ X, const ushortT* __restrict__ Wt,
    const float* __restrict__ bias, float* __restrict__ Out, int M)
{
    constexpr int B_LD   = K + 8;    // B LDS row stride: (K+8)*2B, 16B aligned, bank-skewed
    constexpr int KSTEPS = K / 32;
    __shared__ __align__(16) ushortT bLds[128 * B_LD];
    __shared__ __align__(16) ushortT aLds[2][128 * A_LD];
    __shared__ float ldsCol[128];

    int tid = threadIdx.x, lane = tid & 31, wave = tid >> 5;
    long rowBase = (long)blockIdx.x * 128;

    // ---- async-stage the full B panel (128 x K bf16) Global -> LDS (ASYNCcnt path)
    constexpr int NB = 128 * K / 8;  // number of b128 transfers
    for (int i = tid; i < NB; i += 256) {
        int r = i / (K / 8), c8 = i % (K / 8);
        unsigned ldsOff = (unsigned)(size_t)&bLds[r * B_LD + c8 * 8];
        const ushortT* gp = Wt + r * K + c8 * 8;
        asm volatile("global_load_async_to_lds_b128 %0, %1, off"
                     :: "v"(ldsOff), "v"(gp) : "memory");
    }
    if (TANHSUM && tid < 128) ldsCol[tid] = 0.f;

    // ---- prefetch first A tile (128 x 32 f32) into registers
    float4 xr[4];
    auto loadX = [&](int ks) {
        int kbase = ks * 32;
        #pragma unroll
        for (int i = 0; i < 4; ++i) {
            int f = tid + i * 256;               // float4 id 0..1023
            int r = f >> 3, c4 = (f & 7) * 4;
            long gr = rowBase + r;
            xr[i] = make_float4(0.f, 0.f, 0.f, 0.f);
            if (gr < M) xr[i] = *(const float4*)(X + gr * K + kbase + c4);
        }
    };
    loadX(0);

    asm volatile("s_wait_asynccnt 0x0" ::: "memory");   // own async B copies done

    v8f acc[8] = {};
    int buf = 0;
    for (int ks = 0; ks < KSTEPS; ++ks) {
        // convert + store A tile regs -> LDS[buf] (packed ds_store_b64)
        #pragma unroll
        for (int i = 0; i < 4; ++i) {
            int f = tid + i * 256;
            int r = f >> 3, c4 = (f & 7) * 4;
            uint2 p;
            p.x = pk2bf(xr[i].x, xr[i].y);
            p.y = pk2bf(xr[i].z, xr[i].w);
            *(uint2*)&aLds[buf][r * A_LD + c4] = p;
        }
        __syncthreads();                          // B ready (1st iter); A[buf] visible
        if (ks + 1 < KSTEPS) loadX(ks + 1);       // overlap next X tile with WMMAs

        int rr = lane & 15, kg = lane >> 4;
        const ushortT* ab = &aLds[buf][(wave * 16 + rr) * A_LD];
        FragB16 a;  // A 16x32 bf16: elems0-7 = K kg*8+{0..7}, elems8-15 = K 16+kg*8+{0..7}
        a.q[0] = *(const uint4*)(ab + kg * 8);
        a.q[1] = *(const uint4*)(ab + 16 + kg * 8);
        int kbase = ks * 32;
        #pragma unroll
        for (int nt = 0; nt < 8; ++nt) {
            int n = nt * 16 + rr;
            FragB16 b; // B 32x16 bf16: elems0-7 = K kg*8+{0..7}, elems8-15 = K 16+kg*8+{0..7} of column n
            const uint4* bp = (const uint4*)&bLds[n * B_LD + kbase + kg * 8];
            b.q[0] = bp[0]; b.q[1] = bp[2];
            acc[nt] = __builtin_amdgcn_wmma_f32_16x16x32_bf16(
                false, a.v, false, b.v, (short)0, acc[nt], false, false);
        }
        buf ^= 1;
    }

    // ---- epilogue (C/D layout: lane<16 -> M=vgpr r, lane>=16 -> M=8+r; N = nt*16 + lane%16)
    int rr = lane & 15, hw = lane >> 4;
    if (!TANHSUM) {
        #pragma unroll
        for (int nt = 0; nt < 8; ++nt) {
            int col = nt * 16 + rr;
            float bb = bias[col];
            #pragma unroll
            for (int i = 0; i < 8; ++i) {
                long m = rowBase + wave * 16 + hw * 8 + i;
                if (m < M) Out[m * 128 + col] = acc[nt][i] + bb;
            }
        }
    } else {
        #pragma unroll
        for (int nt = 0; nt < 8; ++nt) {
            int col = nt * 16 + rr;
            float bb = bias[col], s = 0.f;
            #pragma unroll
            for (int i = 0; i < 8; ++i) {
                long m = rowBase + wave * 16 + hw * 8 + i;
                if (m < M) s += tanhf(acc[nt][i] + bb);
            }
            atomicAdd(&ldsCol[col], s);           // ds_add_f32
        }
        __syncthreads();
        if (tid < 128) atomicAdd(&Out[tid], ldsCol[tid]);
    }
}

// ---------------------------------------------------------------- per-node attention scores
__global__ void han_scores(const float* __restrict__ h,
                           const float* a0, const float* a1, const float* a2, const float* a3,
                           float* __restrict__ sc, int N, int nvec)
{
    int idx = blockIdx.x * blockDim.x + threadIdx.x;
    if (idx >= N * 8) return;
    int n = idx >> 3, head = idx & 7;
    const float* hrow = h + (long)n * 128 + head * 16;
    const float* av[4] = { a0, a1, a2, a3 };
    for (int v = 0; v < nvec; ++v) {
        const float* at = av[v] + head * 16;
        float s = 0.f;
        #pragma unroll
        for (int j = 0; j < 16; ++j) s += hrow[j] * at[j];
        sc[((long)n * nvec + v) * 8 + head] = s;
    }
}

// ---------------------------------------------------------------- edge attention phases
__global__ void han_edge_max(const int* __restrict__ src, const int* __restrict__ dst,
                             const float* __restrict__ es, int ess,
                             const float* __restrict__ ed, int eds,
                             unsigned* __restrict__ mx, int E)
{
    int e = blockIdx.x * blockDim.x + threadIdx.x;
    if (e >= E) return;
    int s = src[e], d = dst[e];
    #pragma unroll
    for (int h = 0; h < 8; ++h) {
        float a = es[(long)s * ess + h] + ed[(long)d * eds + h];
        a = (a > 0.f) ? a : NEG_SLOPE * a;
        atomicMax(&mx[d * 8 + h], encOrd(a));
    }
}

__global__ void han_edge_exp(const int* __restrict__ src, const int* __restrict__ dst,
                             const float* __restrict__ es, int ess,
                             const float* __restrict__ ed, int eds,
                             const unsigned* __restrict__ mx,
                             float* __restrict__ den, float* __restrict__ exb, int E)
{
    int e = blockIdx.x * blockDim.x + threadIdx.x;
    if (e >= E) return;
    int s = src[e], d = dst[e];
    #pragma unroll
    for (int h = 0; h < 8; ++h) {
        float a = es[(long)s * ess + h] + ed[(long)d * eds + h];
        a = (a > 0.f) ? a : NEG_SLOPE * a;
        float ex = __expf(a - decOrd(mx[d * 8 + h]));
        exb[(long)e * 8 + h] = ex;
        atomicAdd(&den[d * 8 + h], ex);
    }
}

// one wave32 per edge: lane covers 4 contiguous columns (same head)
__global__ __launch_bounds__(256) void han_edge_aggr(
    const int* __restrict__ src, const int* __restrict__ dst,
    const float* __restrict__ hsrc, const float* __restrict__ den,
    const float* __restrict__ exb, float* __restrict__ out, int E)
{
    int e = blockIdx.x * 8 + (threadIdx.x >> 5);
    if (e >= E) return;
    int lane = threadIdx.x & 31;
    int s = src[e], d = dst[e];
    int head = lane >> 2;
    float w = exb[(long)e * 8 + head] / (den[d * 8 + head] + 1e-16f);
    int c = lane * 4;
    float4 hv = *(const float4*)(hsrc + (long)s * 128 + c);
    float* o = out + (long)d * 128 + c;
    atomicAdd(o + 0, w * hv.x);
    atomicAdd(o + 1, w * hv.y);
    atomicAdd(o + 2, w * hv.z);
    atomicAdd(o + 3, w * hv.w);
}

__global__ void han_relu(float* p, long n) {
    long i = (long)blockIdx.x * blockDim.x + threadIdx.x;
    if (i < n) { float v = p[i]; p[i] = v > 0.f ? v : 0.f; }
}

// ---------------------------------------------------------------- semantic softmax (R=2)
__global__ void han_semsoftmax(const float* __restrict__ colsum, const float* __restrict__ q,
                               float invN, float* __restrict__ attn)
{
    if (threadIdx.x != 0 || blockIdx.x != 0) return;
    float s0 = 0.f, s1 = 0.f;
    for (int i = 0; i < 128; ++i) {
        s0 += q[i] * colsum[i] * invN;
        s1 += q[i] * colsum[128 + i] * invN;
    }
    float m = fmaxf(s0, s1);
    float e0 = __expf(s0 - m), e1 = __expf(s1 - m);
    attn[0] = e0 / (e0 + e1);
    attn[1] = e1 / (e0 + e1);
}

__global__ void han_blend(const float* __restrict__ r0, const float* __restrict__ r1,
                          const float* __restrict__ attn, float* __restrict__ out, int N)
{
    int idx = blockIdx.x * blockDim.x + threadIdx.x;
    if (idx >= N * 32) return;
    int m = idx >> 5, c = (idx & 31) * 4;
    float a0 = attn[0], a1 = attn[1];
    float4 v0 = *(const float4*)(r0 + (long)m * 128 + c);
    float4 v1 = *(const float4*)(r1 + (long)m * 128 + c);
    float4 o;
    o.x = a0 * v0.x + a1 * v1.x; o.y = a0 * v0.y + a1 * v1.y;
    o.z = a0 * v0.z + a1 * v1.z; o.w = a0 * v0.w + a1 * v1.w;
    *(float4*)(out + (long)m * 128 + c) = o;
}

// ---------------------------------------------------------------- classifier + log_softmax
__global__ __launch_bounds__(256) void han_classify(
    const float* __restrict__ X, const float* __restrict__ Wl,
    const float* __restrict__ bl, float* __restrict__ out, int N)
{
    __shared__ float wlS[128 * NCLS];
    __shared__ float blS[NCLS];
    int tid = threadIdx.x;
    for (int i = tid; i < 128 * NCLS; i += 256) wlS[i] = Wl[i];
    if (tid < NCLS) blS[tid] = bl[tid];
    __syncthreads();
    int row = blockIdx.x * 256 + tid;
    if (row >= N) return;
    float acc[NCLS];
    #pragma unroll
    for (int c = 0; c < NCLS; ++c) acc[c] = blS[c];
    const float* x = X + (long)row * 128;
    for (int k = 0; k < 128; ++k) {
        float xv = x[k];
        #pragma unroll
        for (int c = 0; c < NCLS; ++c) acc[c] += xv * wlS[k * NCLS + c];
    }
    float m = acc[0];
    #pragma unroll
    for (int c = 1; c < NCLS; ++c) m = fmaxf(m, acc[c]);
    float s = 0.f;
    #pragma unroll
    for (int c = 0; c < NCLS; ++c) s += __expf(acc[c] - m);
    float l = logf(s);
    #pragma unroll
    for (int c = 0; c < NCLS; ++c) out[(long)row * NCLS + c] = acc[c] - m - l;
}

// ---------------------------------------------------------------- host orchestration
static inline int ceilDiv(long a, long b) { return (int)((a + b - 1) / b); }

extern "C" void kernel_launch(void* const* d_in, const int* in_sizes, int n_in,
                              void* d_out, int out_size, void* d_ws, size_t ws_size,
                              hipStream_t stream)
{
    (void)in_sizes; (void)n_in; (void)out_size; (void)ws_size;
    const float* x_m = (const float*)d_in[0];
    const float* x_d = (const float*)d_in[1];
    const float* x_a = (const float*)d_in[2];
    const float* W_m = (const float*)d_in[3];  const float* b_m = (const float*)d_in[4];
    const float* W_d = (const float*)d_in[5];  const float* b_d = (const float*)d_in[6];
    const float* W_a = (const float*)d_in[7];  const float* b_a = (const float*)d_in[8];
    const float* as_md = (const float*)d_in[9];  const float* ad_md = (const float*)d_in[10];
    const float* as_dm = (const float*)d_in[11]; const float* ad_dm = (const float*)d_in[12];
    const float* as_ma = (const float*)d_in[13]; const float* ad_ma = (const float*)d_in[14];
    const float* as_am = (const float*)d_in[15]; const float* ad_am = (const float*)d_in[16];
    const float* q  = (const float*)d_in[17];
    const float* Wk = (const float*)d_in[18]; const float* bk = (const float*)d_in[19];
    const float* Wl = (const float*)d_in[20]; const float* bl = (const float*)d_in[21];
    const int* md_s = (const int*)d_in[22]; const int* md_d = (const int*)d_in[23];
    const int* dm_s = (const int*)d_in[24]; const int* dm_d = (const int*)d_in[25];
    const int* ma_s = (const int*)d_in[26]; const int* ma_d = (const int*)d_in[27];
    const int* am_s = (const int*)d_in[28]; const int* am_d = (const int*)d_in[29];

    float* outF = (float*)d_out;
    const long o_ls  = 0;                      // log_softmax  [NM][16]
    const long o_mov = (long)NM * NCLS;        // out_movie    [NM][128]
    const long o_dir = o_mov + (long)NM * 128; // out_director [ND][128]
    const long o_act = o_dir + (long)ND * 128; // out_actor    [NA][128]

    // workspace carve-out
    char* ws = (char*)d_ws;
    size_t off = 0;
    auto carve = [&](size_t bytes) -> char* {
        char* p = ws + off;
        off = (off + bytes + 255) & ~(size_t)255;
        return p;
    };
    ushortT* wtM = (ushortT*)carve(128 * 256 * 2);
    ushortT* wtD = (ushortT*)carve(128 * 256 * 2);
    ushortT* wtA = (ushortT*)carve(128 * 256 * 2);
    ushortT* wkt = (ushortT*)carve(128 * 128 * 2);
    float* h_m  = (float*)carve((size_t)NM * 128 * 4);
    float* h_d  = (float*)carve((size_t)ND * 128 * 4);
    float* h_a  = (float*)carve((size_t)NA * 128 * 4);
    float* sc_m = (float*)carve((size_t)NM * 4 * 8 * 4);
    float* sc_d = (float*)carve((size_t)ND * 2 * 8 * 4);
    float* sc_a = (float*)carve((size_t)NA * 2 * 8 * 4);
    float* mx   = (float*)carve((size_t)NA * 8 * 4);   // reused per relation
    float* den  = (float*)carve((size_t)NA * 8 * 4);
    float* exb  = (float*)carve((size_t)E_MA * 8 * 4); // reused per relation
    float* r_dm = (float*)carve((size_t)NM * 128 * 4);
    float* r_am = (float*)carve((size_t)NM * 128 * 4);
    float* colsum = (float*)carve(2 * 128 * 4);
    float* attn   = (float*)carve(2 * 4);

    // 1) weight transpose + bf16 conversion
    han_transcvt<<<ceilDiv(256 * 128, 256), 256, 0, stream>>>(W_m, wtM, 256, 128);
    han_transcvt<<<ceilDiv(256 * 128, 256), 256, 0, stream>>>(W_d, wtD, 256, 128);
    han_transcvt<<<ceilDiv(256 * 128, 256), 256, 0, stream>>>(W_a, wtA, 256, 128);
    han_transcvt<<<ceilDiv(128 * 128, 256), 256, 0, stream>>>(Wk, wkt, 128, 128);

    // 2) WMMA projections h = x @ W + b (async-staged B panel, double-buffered A)
    han_gemm<256, false><<<ceilDiv(NM, 128), 256, 0, stream>>>(x_m, wtM, b_m, h_m, NM);
    han_gemm<256, false><<<ceilDiv(ND, 128), 256, 0, stream>>>(x_d, wtD, b_d, h_d, ND);
    han_gemm<256, false><<<ceilDiv(NA, 128), 256, 0, stream>>>(x_a, wtA, b_a, h_a, NA);

    // 3) per-node attention scores
    //    sc_m: v0=src_md v1=src_ma v2=dst_dm v3=dst_am ; sc_d: v0=dst_md v1=src_dm ; sc_a: v0=dst_ma v1=src_am
    han_scores<<<ceilDiv((long)NM * 8, 256), 256, 0, stream>>>(h_m, as_md, as_ma, ad_dm, ad_am, sc_m, NM, 4);
    han_scores<<<ceilDiv((long)ND * 8, 256), 256, 0, stream>>>(h_d, ad_md, as_dm, nullptr, nullptr, sc_d, ND, 2);
    han_scores<<<ceilDiv((long)NA * 8, 256), 256, 0, stream>>>(h_a, ad_ma, as_am, nullptr, nullptr, sc_a, NA, 2);

    // 4) four relations
    struct Rel {
        const int *s, *d; const float *es; int ess; const float *ed; int eds;
        int E, ndst; const float* hsrc; float* out;
    };
    Rel rels[4] = {
        { md_s, md_d, sc_m + 0,  32, sc_d + 0,  16, E_MD, ND, h_m, outF + o_dir }, // md -> director
        { dm_s, dm_d, sc_d + 8,  16, sc_m + 16, 32, E_MD, NM, h_d, r_dm },         // dm -> movie
        { ma_s, ma_d, sc_m + 8,  32, sc_a + 0,  16, E_MA, NA, h_m, outF + o_act }, // ma -> actor
        { am_s, am_d, sc_a + 8,  16, sc_m + 24, 32, E_MA, NM, h_a, r_am },         // am -> movie
    };
    for (int ri = 0; ri < 4; ++ri) {
        Rel& R = rels[ri];
        long nMH = (long)R.ndst * 8, nOut = (long)R.ndst * 128;
        han_fill_f32<<<ceilDiv(nMH, 256), 256, 0, stream>>>(mx, 0.0f, nMH);  // bits0 < encOrd(-inf)
        han_fill_f32<<<ceilDiv(nMH, 256), 256, 0, stream>>>(den, 0.0f, nMH);
        han_fill_f32<<<ceilDiv(nOut, 256), 256, 0, stream>>>(R.out, 0.0f, nOut);
        han_edge_max<<<ceilDiv(R.E, 256), 256, 0, stream>>>(R.s, R.d, R.es, R.ess, R.ed, R.eds,
                                                            (unsigned*)mx, R.E);
        han_edge_exp<<<ceilDiv(R.E, 256), 256, 0, stream>>>(R.s, R.d, R.es, R.ess, R.ed, R.eds,
                                                            (const unsigned*)mx, den, exb, R.E);
        han_edge_aggr<<<ceilDiv(R.E, 8), 256, 0, stream>>>(R.s, R.d, R.hsrc, den, exb, R.out, R.E);
        han_relu<<<ceilDiv(nOut, 256), 256, 0, stream>>>(R.out, nOut);
    }

    // 5) semantic attention over {r_dm, r_am} (fused WMMA tanh column-sum)
    han_fill_f32<<<1, 256, 0, stream>>>(colsum, 0.0f, 256);
    han_gemm<128, true><<<ceilDiv(NM, 128), 256, 0, stream>>>(r_dm, wkt, bk, colsum + 0, NM);
    han_gemm<128, true><<<ceilDiv(NM, 128), 256, 0, stream>>>(r_am, wkt, bk, colsum + 128, NM);
    han_semsoftmax<<<1, 1, 0, stream>>>(colsum, q, 1.0f / (float)NM, attn);
    han_blend<<<ceilDiv((long)NM * 32, 256), 256, 0, stream>>>(r_dm, r_am, attn, outF + o_mov, NM);

    // 6) classifier + log_softmax
    han_classify<<<ceilDiv(NM, 256), 256, 0, stream>>>(outF + o_mov, Wl, bl, outF + o_ls, NM);
}
